// MultiHeadAttentionLayer_47991964566324
// MI455X (gfx1250) — matrix operand order repaired and checked by reading, not compiled, not run on hardware
//
#include <hip/hip_runtime.h>
#include <stdint.h>

// ---------------------------------------------------------------------------
// MI455X (gfx1250) multi-head attention with ALiBi, wave32 + WMMA bf16.
//   B=2, S=2048, HID=1024, NH=16, HD=64
// d_out = [ x : B*S*HID f32 ][ attention : B*NH*S*S f32 ]
// d_ws  = Qh | Kh | Vt | Xatt (bf16, 8 MB each) | Wq,Wk,Wv,Wo bf16 (2 MB each)
// ---------------------------------------------------------------------------

typedef __attribute__((ext_vector_type(16))) __bf16 v16bf;
typedef __attribute__((ext_vector_type(8)))  __bf16 v8bf;
typedef __attribute__((ext_vector_type(8)))  float  v8f;
typedef __attribute__((ext_vector_type(4)))  float  v4f;

constexpr int B_   = 2;
constexpr int S_   = 2048;
constexpr int HID_ = 1024;
constexpr int NH_  = 16;
constexpr int HD_  = 64;

#define WMMA_BF16(A, Bm, C) \
  __builtin_amdgcn_wmma_f32_16x16x32_bf16(false, (A), false, (Bm), (short)0, (C), false, false)

// 8 contiguous floats at pa -> [0..7], 8 at pb -> [8..15], converted to bf16.
__device__ inline v16bf bf16_from_2x8f(const float* pa, const float* pb) {
  v4f a0 = *(const v4f*)pa, a1 = *(const v4f*)(pa + 4);
  v4f b0 = *(const v4f*)pb, b1 = *(const v4f*)(pb + 4);
  v16bf r;
#pragma unroll
  for (int i = 0; i < 4; ++i) {
    r[i]      = (__bf16)a0[i];
    r[4 + i]  = (__bf16)a1[i];
    r[8 + i]  = (__bf16)b0[i];
    r[12 + i] = (__bf16)b1[i];
  }
  return r;
}

// 8 contiguous bf16 at pa -> [0..7], 8 at pb -> [8..15].
__device__ inline v16bf bf16_from_2x8bf(const __bf16* pa, const __bf16* pb) {
  v8bf a = *(const v8bf*)pa, b = *(const v8bf*)pb;
  v16bf r;
#pragma unroll
  for (int i = 0; i < 8; ++i) { r[i] = a[i]; r[8 + i] = b[i]; }
  return r;
}

// Async 16-byte global -> LDS copy (GLOBAL_LOAD_ASYNC_TO_LDS_B128, GV mode).
// VDST = VGPR with LDS byte address (low 32 bits of generic LDS pointer,
// per ISA 10.2 aperture truncation); VADDR = 64-bit global address.
__device__ inline void async_copy_b128_to_lds(void* lds_ptr, const void* gptr) {
  uint32_t lds_addr = (uint32_t)(uintptr_t)lds_ptr;
  uint64_t gaddr = (uint64_t)(uintptr_t)gptr;
  asm volatile("global_load_async_to_lds_b128 %0, %1, off"
               :: "v"(lds_addr), "v"(gaddr) : "memory");
}
__device__ inline void wait_asynccnt0() {
  asm volatile("s_wait_asynccnt 0x0" ::: "memory");
}

// ---------------------------------------------------------------------------
// One-time fp32 -> bf16 conversion (weights). 8 elems/thread, n % 8 == 0.
// ---------------------------------------------------------------------------
__global__ __launch_bounds__(256) void cvt_f32_to_bf16(
    const float* __restrict__ src, __bf16* __restrict__ dst, int n) {
  const int i = (blockIdx.x * 256 + threadIdx.x) * 8;
  if (i < n) {
    v4f a = *(const v4f*)(src + i);
    v4f b = *(const v4f*)(src + i + 4);
    v8bf o;
#pragma unroll
    for (int t = 0; t < 4; ++t) { o[t] = (__bf16)a[t]; o[4 + t] = (__bf16)b[t]; }
    *(v8bf*)(dst + i) = o;
  }
}

// ---------------------------------------------------------------------------
// GEMM: Y = X @ W^T + bias.  X: [4096, 1024], W (bf16): [1024, 1024] row-major.
// WG tile 64(M) x 128(N); 8 waves, each wave -> 64x16 (4 accumulators).
// Per 32-wide k-chunk: A panel 64x32 staged in LDS (bf16) by all threads,
// one B-fragment per wave reused across 4 WMMAs.
// MODE 0: X=f32, Y=bf16 Qh[b,h,s,d] * 0.125 (fold 1/sqrt(HD))
// MODE 1: X=f32, Y=bf16 Kh[b,h,s,d]
// MODE 2: X=f32, Y=bf16 Vt[b,h,d,s]   (transposed for P@V B-fragments)
// MODE 3: X=bf16 (Xatt), Y=f32 d_out  (A panel staged via async-to-LDS DMA)
// ---------------------------------------------------------------------------
template <int MODE>
__global__ __launch_bounds__(256) void proj_gemm(
    const float* __restrict__ Xf, const __bf16* __restrict__ Xb,
    const __bf16* __restrict__ Wbf, const float* __restrict__ bias,
    __bf16* __restrict__ Yb, float* __restrict__ Yf) {
  __shared__ __bf16 As[64][40];  // 64 rows x 32 k (+8 pad), 16B-aligned chunks

  const int lane = threadIdx.x & 31;
  const int wave = threadIdx.x >> 5;
  const int half = lane >> 4;
  const int m0 = blockIdx.x * 64;             // WG M origin (0..4032)
  const int ntile = blockIdx.y * 8 + wave;    // 0..63
  const int wrow = ntile * 16 + (lane & 15);  // W row == output column

  const int sr = threadIdx.x >> 2;        // staging row 0..63
  const int sc = (threadIdx.x & 3) * 8;   // staging col 0,8,16,24

  v8f acc[4];
#pragma unroll
  for (int t = 0; t < 4; ++t) acc[t] = (v8f){0.f,0.f,0.f,0.f,0.f,0.f,0.f,0.f};

  for (int kc = 0; kc < HID_; kc += 32) {
    // ---- stage 64x32 A panel into LDS as bf16
    if (MODE == 3) {
      // bf16 source: pure copy -> async DMA straight into LDS (ASYNCcnt path)
      async_copy_b128_to_lds(&As[sr][sc],
                             Xb + (size_t)(m0 + sr) * HID_ + kc + sc);
      wait_asynccnt0();
    } else {
      const float* p = Xf + (size_t)(m0 + sr) * HID_ + kc + sc;
      v4f a = *(const v4f*)p, b = *(const v4f*)(p + 4);
      v8bf o;
#pragma unroll
      for (int t = 0; t < 4; ++t) { o[t] = (__bf16)a[t]; o[4 + t] = (__bf16)b[t]; }
      *(v8bf*)&As[sr][sc] = o;
    }
    __syncthreads();

    // ---- one B-fragment per wave, reused by 4 WMMAs
    const __bf16* pb = Wbf + (size_t)wrow * HID_ + kc + half * 8;
    __builtin_prefetch((const void*)(pb + 128), 0, 0);  // global_prefetch_b8
    const v16bf bfrag = bf16_from_2x8bf(pb, pb + 16);
#pragma unroll
    for (int t = 0; t < 4; ++t) {
      const __bf16* pa = &As[t * 16 + (lane & 15)][half * 8];
      const v16bf afrag = bf16_from_2x8bf(pa, pa + 16);
      acc[t] = WMMA_BF16(afrag, bfrag, acc[t]);
    }
    __syncthreads();
  }

  const int ncol = wrow;
  const float bvv = bias[ncol];
  const int h = ncol >> 6, d = ncol & 63;
#pragma unroll
  for (int t = 0; t < 4; ++t) {
#pragma unroll
    for (int r = 0; r < 8; ++r) {
      const int grow = m0 + t * 16 + r + half * 8;  // global row in [0, 4096)
      const int b = grow >> 11;                     // / 2048
      const int s = grow & 2047;
      const float val = acc[t][r] + bvv;
      if (MODE == 0) {
        Yb[(((size_t)(b * NH_ + h)) * S_ + s) * HD_ + d] = (__bf16)(val * 0.125f);
      } else if (MODE == 1) {
        Yb[(((size_t)(b * NH_ + h)) * S_ + s) * HD_ + d] = (__bf16)val;
      } else if (MODE == 2) {
        Yb[(((size_t)(b * NH_ + h)) * HD_ + d) * S_ + s] = (__bf16)val;
      } else {
        Yf[(size_t)grow * HID_ + ncol] = val;
      }
    }
  }
}

// ---------------------------------------------------------------------------
// Attention: one WG per (b, h, 16-row query tile). Scores strip 16x2048 f32
// lives in LDS (128 KB of the 320 KB WGP pool). Softmax in-place, probs
// streamed once to d_out, then P@V via WMMA reading P back from LDS.
// ---------------------------------------------------------------------------
__global__ __launch_bounds__(256) void attn_kernel(
    const __bf16* __restrict__ Qh, const __bf16* __restrict__ Kh,
    const __bf16* __restrict__ Vt, const int* __restrict__ mask,
    const float* __restrict__ slopes, float* __restrict__ attn_out,
    __bf16* __restrict__ Xatt) {
  __shared__ float strip[16 * S_];   // 128 KB
  __shared__ float red[16][17];
  __shared__ float rowstat[16];

  const int lane = threadIdx.x & 31;
  const int wave = threadIdx.x >> 5;
  const int half = lane >> 4;
  const int ibase = blockIdx.x * 16;
  const int h = blockIdx.y;
  const int b = blockIdx.z;
  const int bh = b * NH_ + h;
  const float slope = slopes[h];

  // --- Q A-fragments (16x64 -> two K=32 fragments), reused for all j tiles.
  const __bf16* qrow = Qh + ((size_t)bh * S_ + (ibase + (lane & 15))) * HD_ + half * 8;
  const v16bf qa0 = bf16_from_2x8bf(qrow, qrow + 16);
  const v16bf qa1 = bf16_from_2x8bf(qrow + 32, qrow + 48);

  // --- scores: each wave covers 16 of the 128 j-tiles.
  for (int t = 0; t < 16; ++t) {
    const int jbase = (t * 8 + wave) * 16;
    const int j = jbase + (lane & 15);
    const __bf16* krow = Kh + ((size_t)bh * S_ + j) * HD_ + half * 8;
    const v16bf kb0 = bf16_from_2x8bf(krow, krow + 16);
    const v16bf kb1 = bf16_from_2x8bf(krow + 32, krow + 48);
    v8f acc = {0.f, 0.f, 0.f, 0.f, 0.f, 0.f, 0.f, 0.f};
    acc = WMMA_BF16(qa0, kb0, acc);
    acc = WMMA_BF16(qa1, kb1, acc);
    const float alibi = slope * (float)j;          // key-index bias (lane const)
    const bool masked = (mask[b * S_ + j] == 0);
#pragma unroll
    for (int r = 0; r < 8; ++r) {
      float v = acc[r] + alibi;
      if (masked) v = -1e10f;
      strip[(r + half * 8) * S_ + j] = v;
    }
  }
  __syncthreads();

  // --- softmax over each of the 16 rows (2048 wide); 16 threads per row.
  {
    const int r = threadIdx.x >> 4;
    const int seg = threadIdx.x & 15;
    float* rowp = &strip[r * S_ + seg * 128];
    float mx = -3.0e38f;
    for (int i = 0; i < 128; ++i) mx = fmaxf(mx, rowp[i]);
    red[r][seg] = mx;
    __syncthreads();
    if (threadIdx.x < 16) {
      float m = red[threadIdx.x][0];
      for (int i = 1; i < 16; ++i) m = fmaxf(m, red[threadIdx.x][i]);
      rowstat[threadIdx.x] = m;
    }
    __syncthreads();
    const float rmax = rowstat[r];
    float sum = 0.f;
    for (int i = 0; i < 128; ++i) {
      const float e = __expf(rowp[i] - rmax);
      rowp[i] = e;
      sum += e;
    }
    red[r][seg] = sum;
    __syncthreads();
    if (threadIdx.x < 16) {
      float s2 = 0.f;
      for (int i = 0; i < 16; ++i) s2 += red[threadIdx.x][i];
      rowstat[threadIdx.x] = 1.0f / s2;
    }
    __syncthreads();
    const float rinv = rowstat[r];
    float* gout = attn_out + ((size_t)bh * S_ + (ibase + r)) * S_ + seg * 128;
    for (int i = 0; i < 128; ++i) {
      const float p = rowp[i] * rinv;
      rowp[i] = p;   // keep normalized probs in LDS for P@V
      gout[i] = p;   // the attention output (f32)
    }
  }
  __syncthreads();

  // --- context: O[16 x 64] = P[16 x 2048] @ V[2048 x 64]; waves 0..3, one
  //     16-wide d-tile each, 64 K=32 WMMAs. B-fragments contiguous via Vt.
  if (wave < 4) {
    const int dcol = wave * 16 + (lane & 15);
    const __bf16* vrow = Vt + ((size_t)bh * HD_ + dcol) * S_;
    v8f acc = {0.f, 0.f, 0.f, 0.f, 0.f, 0.f, 0.f, 0.f};
    for (int kk = 0; kk < 64; ++kk) {
      const float* pa = &strip[(lane & 15) * S_ + kk * 32 + half * 8];
      const v16bf af = bf16_from_2x8f(pa, pa + 16);     // ds_load + cvt
      const __bf16* pb = vrow + kk * 32 + half * 8;
      const v16bf bf_ = bf16_from_2x8bf(pb, pb + 16);
      acc = WMMA_BF16(af, bf_, acc);
    }
#pragma unroll
    for (int r = 0; r < 8; ++r) {
      const int srow = ibase + r + half * 8;
      Xatt[((size_t)b * S_ + srow) * HID_ + h * HD_ + dcol] = (__bf16)acc[r];
    }
  }
}

// ---------------------------------------------------------------------------
extern "C" void kernel_launch(void* const* d_in, const int* in_sizes, int n_in,
                              void* d_out, int out_size, void* d_ws, size_t ws_size,
                              hipStream_t stream) {
  const float* query = (const float*)d_in[0];
  const float* key_  = (const float*)d_in[1];
  const float* value = (const float*)d_in[2];
  const int*   mask  = (const int*)d_in[3];
  const float* Wq = (const float*)d_in[4];
  const float* bq = (const float*)d_in[5];
  const float* Wk = (const float*)d_in[6];
  const float* bk = (const float*)d_in[7];
  const float* Wv = (const float*)d_in[8];
  const float* bv = (const float*)d_in[9];
  const float* Wo = (const float*)d_in[10];
  const float* bo = (const float*)d_in[11];
  const float* slopes = (const float*)d_in[12];

  const size_t NELEM = (size_t)B_ * S_ * HID_;   // 4194304
  const size_t WELEM = (size_t)HID_ * HID_;      // 1048576
  __bf16* Qh   = (__bf16*)d_ws;
  __bf16* Kh   = Qh + NELEM;
  __bf16* Vt   = Kh + NELEM;
  __bf16* Xatt = Vt + NELEM;
  __bf16* Wqb  = Xatt + NELEM;
  __bf16* Wkb  = Wqb + WELEM;
  __bf16* Wvb  = Wkb + WELEM;
  __bf16* Wob  = Wvb + WELEM;

  float* Xout = (float*)d_out;
  float* Attn = Xout + NELEM;

  dim3 blk(256, 1, 1);

  // one-time bf16 weight conversion
  const int cgrid = (int)(WELEM / 8 / 256);  // 512
  cvt_f32_to_bf16<<<cgrid, blk, 0, stream>>>(Wq, Wqb, (int)WELEM);
  cvt_f32_to_bf16<<<cgrid, blk, 0, stream>>>(Wk, Wkb, (int)WELEM);
  cvt_f32_to_bf16<<<cgrid, blk, 0, stream>>>(Wv, Wvb, (int)WELEM);
  cvt_f32_to_bf16<<<cgrid, blk, 0, stream>>>(Wo, Wob, (int)WELEM);

  dim3 ggrid((B_ * S_) / 64, HID_ / 128, 1);  // 64 x 8

  proj_gemm<0><<<ggrid, blk, 0, stream>>>(query, nullptr, Wqb, bq, Qh, nullptr);
  proj_gemm<1><<<ggrid, blk, 0, stream>>>(key_,  nullptr, Wkb, bk, Kh, nullptr);
  proj_gemm<2><<<ggrid, blk, 0, stream>>>(value, nullptr, Wvb, bv, Vt, nullptr);

  dim3 agrid(S_ / 16, NH_, B_);  // 128 x 16 x 2 workgroups
  attn_kernel<<<agrid, blk, 0, stream>>>(Qh, Kh, Vt, mask, slopes, Attn, Xatt);

  proj_gemm<3><<<ggrid, blk, 0, stream>>>(nullptr, Xatt, Wob, bo, nullptr, Xout);
}
